// MultiHeadAttention_35364760715489
// MI455X (gfx1250) — matrix operand order repaired and checked by reading, not compiled, not run on hardware
//
#include <hip/hip_runtime.h>
#include <hip/hip_bf16.h>

// MI455X (gfx1250) multi-head attention, bf16 WMMA + f32 accumulate.
// D=1024, H=16, d_k=64, B=4, S=2048.

typedef __bf16 bf16;
typedef __attribute__((ext_vector_type(16))) __bf16 v16bf;
typedef __attribute__((ext_vector_type(8)))  __bf16 v8bf;
typedef __attribute__((ext_vector_type(4)))  __bf16 v4bf;
typedef __attribute__((ext_vector_type(8)))  float  v8f;
typedef __attribute__((ext_vector_type(4)))  float  v4f;
typedef __attribute__((ext_vector_type(4)))  int    v4i;

// 16-byte global -> LDS copy. Prefer the gfx1250 async-to-LDS engine
// (ASYNCcnt-tracked, no VGPR round trip); fall back to a sync copy.
static __device__ __forceinline__ void copy16_g2l(const bf16* g, bf16* l) {
#if __has_builtin(__builtin_amdgcn_global_load_async_to_lds_b128)
  typedef __attribute__((address_space(1))) v4i gv4i_t;  // global (AS1)
  typedef __attribute__((address_space(3))) v4i lv4i_t;  // LDS (AS3)
  __builtin_amdgcn_global_load_async_to_lds_b128((gv4i_t*)g, (lv4i_t*)l, 0, 0);
#else
  *(v8bf*)l = *(const v8bf*)g;
#endif
}

static __device__ __forceinline__ void async_wait0() {
#if __has_builtin(__builtin_amdgcn_global_load_async_to_lds_b128)
#if __has_builtin(__builtin_amdgcn_s_wait_asynccnt)
  __builtin_amdgcn_s_wait_asynccnt(0);
#else
  asm volatile("s_wait_asynccnt 0x0" ::: "memory");
#endif
#endif
}

static __device__ __forceinline__ v8f wmma_bf16(v16bf a, v16bf b, v8f c) {
  // D = A(16x32 bf16) * B(32x16 bf16) + C(16x16 f32)
  return __builtin_amdgcn_wmma_f32_16x16x32_bf16(
      /*neg_a=*/false, a, /*neg_b=*/false, b,
      /*c_mod=*/(short)0, c, /*reuse_a=*/false, /*reuse_b=*/false);
}

// A fragment (16x32, row-major source with leading dim ld elements).
// Layout: lanes 0-15 (M=lane) hold K=0..15; lanes 16-31 hold K=16..31 (ascending).
static __device__ __forceinline__ v16bf load_frag_a(const bf16* base, int ld, int lane) {
  const int m  = lane & 15;
  const int kg = (lane >> 4) << 4;          // 0 or 16
  const bf16* p = base + m * ld + kg;
  v8bf a0 = *(const v8bf*)p;                // K = kg .. kg+7
  v8bf a1 = *(const v8bf*)(p + 8);          // K = kg+8 .. kg+15
  v16bf r;
#pragma unroll
  for (int i = 0; i < 8; ++i) { r[i] = a0[i]; r[i + 8] = a1[i]; }
  return r;
}

// B fragment (32x16) from LDS stored as Bs[n][k] (leading dim ld elements).
// Layout: lanes 0-15 (N=lane) hold K=0..15; lanes 16-31 hold K=16..31 (ascending).
static __device__ __forceinline__ v16bf load_frag_b(const bf16* base, int ld, int lane) {
  const bf16* p = base + (lane & 15) * ld + ((lane >> 4) << 4);
  return *(const v16bf*)p;                  // 32 contiguous bytes
}

// ---------------------------------------------------------------------------
// fp32 -> bf16 cast (vectorized by 4)
// ---------------------------------------------------------------------------
__global__ __launch_bounds__(256) void mha_cast_bf16(const float* __restrict__ in,
                                                     bf16* __restrict__ out, int n4) {
  int i = blockIdx.x * blockDim.x + threadIdx.x;
  if (i < n4) {
    v4f v = ((const v4f*)in)[i];
    v4bf o;
#pragma unroll
    for (int j = 0; j < 4; ++j) o[j] = (bf16)v[j];
    ((v4bf*)out)[i] = o;
  }
}

// ---------------------------------------------------------------------------
// C[M,N] = A[M,K](bf16,row-major) * W[N,K]^T (bf16,row-major)
// Block tile 128x128, BK=64. 256 threads = 8 waves (2m x 4n), wave tile 64x32.
// Double-buffered LDS: async loads of tile kt+1 overlap WMMA compute of tile kt.
// Per K-tile: 2 WMMA k-steps, 16 WMMAs/wave vs 12 fragment loads.
// ---------------------------------------------------------------------------
template <bool OUT_BF16>
__global__ __launch_bounds__(256)
void mha_gemm_xwt(const bf16* __restrict__ A, const bf16* __restrict__ W,
                  float* __restrict__ Cf, bf16* __restrict__ Cb,
                  int M, int N, int K) {
  __shared__ alignas(64) bf16 As[2][128][64];  // 2 x 16 KB
  __shared__ alignas(64) bf16 Bs[2][128][64];  // 2 x 16 KB, Bs[.][n][k] = W[n0+n][kb+k]

  const int t = threadIdx.x, lane = t & 31, w = t >> 5;
  const int wm = w >> 2, wn = w & 3;        // 2 x 4 wave grid
  const int m0 = blockIdx.x * 128, n0 = blockIdx.y * 128;

  v8f acc[4][2] = {};
  const int ktiles = K >> 6;

  auto issue_tile = [&](int kt, int buf) {
    const int kb = kt << 6;
    // 128x64 bf16 tile = 1024 chunks of 8 elems (16B); 4 per thread, each matrix.
#pragma unroll
    for (int c = 0; c < 4; ++c) {
      int idx = t + c * 256;
      int r = idx >> 3, o = (idx & 7) << 3;
      copy16_g2l(&A[(size_t)(m0 + r) * K + kb + o], &As[buf][r][o]);
      copy16_g2l(&W[(size_t)(n0 + r) * K + kb + o], &Bs[buf][r][o]);
    }
  };

  issue_tile(0, 0);

  for (int kt = 0; kt < ktiles; ++kt) {
    const int buf = kt & 1;
    // Each wave drains its own async loads, then the barrier makes every
    // wave's DMA-written LDS visible block-wide (and proves all waves are
    // done reading buf^1 from iteration kt-1, so it is safe to refill).
    async_wait0();
    __syncthreads();
    if (kt + 1 < ktiles) issue_tile(kt + 1, buf ^ 1);

#pragma unroll
    for (int kk = 0; kk < 2; ++kk) {
      v16bf af[4], bf[2];
#pragma unroll
      for (int i = 0; i < 4; ++i)
        af[i] = load_frag_a(&As[buf][wm * 64 + i * 16][kk * 32], 64, lane);
#pragma unroll
      for (int j = 0; j < 2; ++j)
        bf[j] = load_frag_b(&Bs[buf][wn * 32 + j * 16][kk * 32], 64, lane);
#pragma unroll
      for (int i = 0; i < 4; ++i)
#pragma unroll
        for (int j = 0; j < 2; ++j)
          acc[i][j] = wmma_bf16(af[i], bf[j], acc[i][j]);
    }
  }

  // C layout: lane l holds col (l&15), rows (l>>4)*8 + r
  const int cn = lane & 15, mh = lane >> 4;
#pragma unroll
  for (int i = 0; i < 4; ++i) {
#pragma unroll
    for (int j = 0; j < 2; ++j) {
#pragma unroll
      for (int r = 0; r < 8; ++r) {
        int row = m0 + wm * 64 + i * 16 + mh * 8 + r;
        int col = n0 + wn * 32 + j * 16 + cn;
        if (OUT_BF16) Cb[(size_t)row * N + col] = (bf16)acc[i][j][r];
        else          Cf[(size_t)row * N + col] = acc[i][j][r];
      }
    }
  }
}

// ---------------------------------------------------------------------------
// Flash-attention (causal). Grid: (S/64, B*H). Block: 128 threads = 4 waves,
// each wave owns 16 query rows. Key tiles of 64; online softmax in registers.
// ---------------------------------------------------------------------------
__global__ __launch_bounds__(128)
void mha_flash_attn(const bf16* __restrict__ Q, const bf16* __restrict__ Km,
                    const bf16* __restrict__ V, bf16* __restrict__ O) {
  constexpr int S = 2048, D = 1024;
  __shared__ alignas(64) bf16 Qs[64][64];   // [q][d]
  __shared__ alignas(64) bf16 Ks[64][64];   // [key][d]  (== Bs[n=key][k=d])
  __shared__ alignas(64) bf16 Vs[64][64];   // [d][key]  (== Bs[n=d][k=key])
  __shared__ alignas(64) bf16 Ps[64][64];   // [q][key]  (A operand for P@V)

  const int t = threadIdx.x, lane = t & 31, w = t >> 5;
  const int qb = blockIdx.x, bh = blockIdx.y;
  const int b = bh >> 4, h = bh & 15;
  const size_t head_off = (size_t)b * S * D + (size_t)h * 64;

  // Q tile: 64x64 bf16, 512 16B-chunks over 128 threads (async; waited below)
  for (int c = t; c < 512; c += 128) {
    int r = c >> 3, o = (c & 7) << 3;
    copy16_g2l(&Q[head_off + (size_t)(qb * 64 + r) * D + o], &Qs[r][o]);
  }

  const int qw = w * 16;
  const int mh = lane >> 4, cn = lane & 15;
  v8f o_acc[4] = {};
  float m_i[8], l_i[8];
#pragma unroll
  for (int r = 0; r < 8; ++r) { m_i[r] = -1e30f; l_i[r] = 0.f; }
  const float scale = 0.125f;  // 1/sqrt(64)

  const int nkt = qb + 1;  // causal: key tiles 0..qb
  for (int kt = 0; kt < nkt; ++kt) {
    __syncthreads();  // previous iteration's Ks/Vs/Ps reads complete
    for (int c = t; c < 512; c += 128) {
      int r = c >> 3, o = (c & 7) << 3;
      copy16_g2l(&Km[head_off + (size_t)(kt * 64 + r) * D + o], &Ks[r][o]);
      v8bf vv = *(const v8bf*)&V[head_off + (size_t)(kt * 64 + r) * D + o];
#pragma unroll
      for (int i = 0; i < 8; ++i) Vs[o + i][r] = vv[i];  // transpose into [d][key]
    }
    async_wait0();
    __syncthreads();

    // S = Q * K^T  (16x64 per wave): 2 k-steps x 4 n-tiles
    v8f s[4] = {};
#pragma unroll
    for (int kk = 0; kk < 2; ++kk) {
      v16bf aq = load_frag_a(&Qs[qw][kk * 32], 64, lane);
#pragma unroll
      for (int nt = 0; nt < 4; ++nt) {
        v16bf bk = load_frag_b(&Ks[nt * 16][kk * 32], 64, lane);
        s[nt] = wmma_bf16(aq, bk, s[nt]);
      }
    }

    // scale + causal mask
    const int qbase = qb * 64 + qw + mh * 8;   // + r
    const int kbase = kt * 64 + cn;            // + nt*16
#pragma unroll
    for (int nt = 0; nt < 4; ++nt) {
#pragma unroll
      for (int r = 0; r < 8; ++r) {
        float sv = s[nt][r] * scale;
        if (kbase + nt * 16 > qbase + r) sv = -1e9f;
        s[nt][r] = sv;
      }
    }

    // online softmax: rows live in 16-lane groups (cols in lanes)
#pragma unroll
    for (int r = 0; r < 8; ++r) {
      float mx = fmaxf(fmaxf(s[0][r], s[1][r]), fmaxf(s[2][r], s[3][r]));
#pragma unroll
      for (int sh = 1; sh < 16; sh <<= 1) mx = fmaxf(mx, __shfl_xor(mx, sh, 32));
      float nm = fmaxf(m_i[r], mx);
      float rs = 0.f;
#pragma unroll
      for (int nt = 0; nt < 4; ++nt) {
        float p = __expf(s[nt][r] - nm);
        s[nt][r] = p;
        rs += p;
      }
#pragma unroll
      for (int sh = 1; sh < 16; sh <<= 1) rs += __shfl_xor(rs, sh, 32);
      float alpha = __expf(m_i[r] - nm);
      l_i[r] = l_i[r] * alpha + rs;
      m_i[r] = nm;
#pragma unroll
      for (int nt = 0; nt < 4; ++nt) o_acc[nt][r] *= alpha;
    }

    // P -> LDS (C layout -> row-major), then consume as A fragments
#pragma unroll
    for (int nt = 0; nt < 4; ++nt) {
#pragma unroll
      for (int r = 0; r < 8; ++r)
        Ps[qw + mh * 8 + r][nt * 16 + cn] = (bf16)s[nt][r];
    }
    __syncthreads();

    // O += P * V : 2 k-steps (keys) x 4 n-tiles (d)
#pragma unroll
    for (int kk = 0; kk < 2; ++kk) {
      v16bf ap = load_frag_a(&Ps[qw][kk * 32], 64, lane);
#pragma unroll
      for (int nt = 0; nt < 4; ++nt) {
        v16bf bv = load_frag_b(&Vs[nt * 16][kk * 32], 64, lane);
        o_acc[nt] = wmma_bf16(ap, bv, o_acc[nt]);
      }
    }
  }

  // normalize and store bf16 attention output [b][q][h*64+d]
#pragma unroll
  for (int r = 0; r < 8; ++r) {
    float inv = 1.0f / l_i[r];
    int q = qb * 64 + qw + mh * 8 + r;
#pragma unroll
    for (int nt = 0; nt < 4; ++nt) {
      int d = nt * 16 + cn;
      O[head_off + (size_t)q * D + d] = (bf16)(o_acc[nt][r] * inv);
    }
  }
}

// ---------------------------------------------------------------------------
extern "C" void kernel_launch(void* const* d_in, const int* in_sizes, int n_in,
                              void* d_out, int out_size, void* d_ws, size_t ws_size,
                              hipStream_t stream) {
  constexpr int Bn = 4, S = 2048, D = 1024;
  constexpr size_t nX = (size_t)Bn * S * D;  // 8388608
  constexpr size_t nW = (size_t)D * D;       // 1048576

  const float* X  = (const float*)d_in[0];
  const float* Wq = (const float*)d_in[1];
  const float* Wk = (const float*)d_in[2];
  const float* Wv = (const float*)d_in[3];
  const float* Wo = (const float*)d_in[4];
  float* out = (float*)d_out;

  bf16* p   = (bf16*)d_ws;
  bf16* Xbf = p;  p += nX;
  bf16* Wqb = p;  p += nW;
  bf16* Wkb = p;  p += nW;
  bf16* Wvb = p;  p += nW;
  bf16* Wob = p;  p += nW;
  bf16* Qb  = p;  p += nX;
  bf16* Kb  = p;  p += nX;
  bf16* Vb  = p;  p += nX;
  bf16* Ab  = p;  p += nX;   // total ~92.3 MB of workspace

  // casts
  mha_cast_bf16<<<(int)(nX / 4 / 256), 256, 0, stream>>>(X,  Xbf, (int)(nX / 4));
  mha_cast_bf16<<<(int)(nW / 4 / 256), 256, 0, stream>>>(Wq, Wqb, (int)(nW / 4));
  mha_cast_bf16<<<(int)(nW / 4 / 256), 256, 0, stream>>>(Wk, Wkb, (int)(nW / 4));
  mha_cast_bf16<<<(int)(nW / 4 / 256), 256, 0, stream>>>(Wv, Wvb, (int)(nW / 4));
  mha_cast_bf16<<<(int)(nW / 4 / 256), 256, 0, stream>>>(Wo, Wob, (int)(nW / 4));

  // projections: Q/K/V = X @ W^T  (M=8192, N=1024, K=1024)
  dim3 gg(64, 8), gb(256);
  mha_gemm_xwt<true><<<gg, gb, 0, stream>>>(Xbf, Wqb, nullptr, Qb, Bn * S, D, D);
  mha_gemm_xwt<true><<<gg, gb, 0, stream>>>(Xbf, Wkb, nullptr, Kb, Bn * S, D, D);
  mha_gemm_xwt<true><<<gg, gb, 0, stream>>>(Xbf, Wvb, nullptr, Vb, Bn * S, D, D);

  // causal flash attention per (b,h)
  mha_flash_attn<<<dim3(S / 64, Bn * 16), 128, 0, stream>>>(Qb, Kb, Vb, Ab);

  // output projection -> fp32
  mha_gemm_xwt<false><<<gg, gb, 0, stream>>>(Ab, Wob, out, nullptr, Bn * S, D, D);
}
